// PPFNet_29386166239365
// MI455X (gfx1250) — matrix operand order repaired and checked by reading, not compiled, not run on hardware
//
#include <hip/hip_runtime.h>
#include <hip/hip_bf16.h>
#include <math.h>

typedef __attribute__((ext_vector_type(16))) _Float16 v16h;
typedef __attribute__((ext_vector_type(8)))  float    v8f;

#define BGR 8
#define NN  4096
#define KNN 16
#define MM  (BGR * NN)          // 32768 nodes
#define EPN (KNN + 1)           // 17 edges per node (16 nbrs + self loop)
#define EE  (MM * EPN)          // 557056 edges, divisible by 16
#define HID 32
#define NCLS 40
#define WAVES_PER_BLOCK 8
#define NTILES 10               // packed f16 B tiles

// ---------------------------------------------------------------- helpers
__device__ __forceinline__ v16h zero_v16h() {
    v16h v;
#pragma unroll
    for (int t = 0; t < 16; ++t) v[t] = (_Float16)0.0f;
    return v;
}

// Packed B tile fetch: tile laid out [tile][lane][16 halves] = 32B per lane,
// built by pack_weights_kernel in the exact WMMA B-operand register layout.
__device__ __forceinline__ v16h load_Bp(const _Float16* __restrict__ p,
                                        int tile, int lane) {
    return ((const v16h*)p)[tile * 32 + lane];
}

__device__ __forceinline__ v8f bias_c(const float* __restrict__ b, int col) {
    v8f c;
#pragma unroll
    for (int r = 0; r < 8; ++r) c[r] = b[col];
    return c;
}

__device__ __forceinline__ v8f wmma16(v16h a, v16h b, v8f c) {
    return __builtin_amdgcn_wmma_f32_16x16x32_f16(false, a, false, b,
                                                  (short)0, c, false, false);
}

// ---------------------------------------------------------------- weight pack
// B layout: lane<16 -> col=lane+colbase, K={0..7,16..23}; lane>=16 -> col=lane-16+colbase,
// K={8..15,24..31}. Zero-pad K beyond kdim. One 32B row per lane per tile.
__global__ void pack_weights_kernel(const float* __restrict__ w1a,
                                    const float* __restrict__ w1b,
                                    const float* __restrict__ w2a,
                                    const float* __restrict__ w2b,
                                    _Float16* __restrict__ packed) {
    const int t    = threadIdx.x;           // 0..319
    const int tile = t >> 5;
    const int lane = t & 31;
    const float* w; int kdim; int colbase;
    switch (tile) {
        case 0: w = w1a;            kdim = 4;  colbase = 0;  break;
        case 1: w = w1a;            kdim = 4;  colbase = 16; break;
        case 2: w = w1b;            kdim = 32; colbase = 0;  break;
        case 3: w = w1b;            kdim = 32; colbase = 16; break;
        case 4: w = w2a;            kdim = 32; colbase = 0;  break;
        case 5: w = w2a;            kdim = 32; colbase = 16; break;
        case 6: w = w2a + 32 * HID; kdim = 4;  colbase = 0;  break;  // rows 32..35
        case 7: w = w2a + 32 * HID; kdim = 4;  colbase = 16; break;
        case 8: w = w2b;            kdim = 32; colbase = 0;  break;
        default: w = w2b;           kdim = 32; colbase = 16; break;
    }
    const int col = (lane & 15) + colbase;
    const int kb  = (lane < 16) ? 0 : 8;
    _Float16* o = packed + ((size_t)tile * 32 + lane) * 16;
#pragma unroll
    for (int s = 0; s < 16; ++s) {
        const int k = (s < 8) ? kb + s : 16 + kb + (s - 8);
        o[s] = (k < kdim) ? (_Float16)w[k * HID + col] : (_Float16)0.0f;
    }
}

// ---------------------------------------------------------------- kNN
// Block = 256 nodes of one graph. Candidates staged through LDS 256 at a
// time; inner loop is pure VALU + LDS broadcast. Top-16 via replace-worst.
__global__ void knn_kernel(const float* __restrict__ pos, int* __restrict__ nbr) {
    __shared__ float sp[256][3];
    const int i    = blockIdx.x * 256 + threadIdx.x;
    const int base = (i >> 12) << 12;                  // graph base node
    const float px = pos[3 * i], py = pos[3 * i + 1], pz = pos[3 * i + 2];
    float d[KNN];
    int   id[KNN];
#pragma unroll
    for (int s = 0; s < KNN; ++s) { d[s] = __builtin_inff(); id[s] = base; }
    float worst = __builtin_inff();
    int   wslot = 0;
    for (int tile = 0; tile < NN; tile += 256) {
        const int j = base + tile + threadIdx.x;
        sp[threadIdx.x][0] = pos[3 * j];
        sp[threadIdx.x][1] = pos[3 * j + 1];
        sp[threadIdx.x][2] = pos[3 * j + 2];
        __syncthreads();
        for (int q = 0; q < 256; ++q) {
            const int jj = base + tile + q;
            const float dx = sp[q][0] - px;
            const float dy = sp[q][1] - py;
            const float dz = sp[q][2] - pz;
            float d2 = dx * dx + dy * dy + dz * dz;
            if (jj == i) d2 = __builtin_inff();        // exclude self
            if (d2 < worst) {
#pragma unroll
                for (int s = 0; s < KNN; ++s)
                    if (s == wslot) { d[s] = d2; id[s] = jj; }
                worst = -__builtin_inff();
#pragma unroll
                for (int s = 0; s < KNN; ++s)
                    if (d[s] > worst) { worst = d[s]; wslot = s; }
            }
        }
        __syncthreads();
    }
#pragma unroll
    for (int s = 0; s < KNN; ++s) nbr[i * KNN + s] = id[s];
}

// ---------------------------------------------------------------- PPF feats
__device__ __forceinline__ float safe_norm3(float x, float y, float z) {
    const float s = x * x + y * y + z * z;
    return s > 0.0f ? sqrtf(s) : 0.0f;
}
__device__ __forceinline__ float angle3(float ax, float ay, float az,
                                        float bx, float by, float bz) {
    const float cx = ay * bz - az * by;
    const float cy = az * bx - ax * bz;
    const float cz = ax * by - ay * bx;
    const float cn = safe_norm3(cx, cy, cz);
    const float dd = ax * bx + ay * by + az * bz;
    const bool ok  = (cn > 0.0f) || (dd != 0.0f);
    return ok ? atan2f(cn, dd) : 0.0f;
}

__global__ void ppf_kernel(const float* __restrict__ pos,
                           const float* __restrict__ nrm,
                           const int* __restrict__ nbr,
                           float* __restrict__ feats) {
    const int e = blockIdx.x * blockDim.x + threadIdx.x;
    if (e >= EE) return;
    const int i = e / EPN;
    const int k = e - i * EPN;
    const int j = (k < KNN) ? nbr[i * KNN + k] : i;   // src; dst = i
    const float px = pos[3 * j] - pos[3 * i];
    const float py = pos[3 * j + 1] - pos[3 * i + 1];
    const float pz = pos[3 * j + 2] - pos[3 * i + 2];
    const float nix = nrm[3 * i], niy = nrm[3 * i + 1], niz = nrm[3 * i + 2];
    const float njx = nrm[3 * j], njy = nrm[3 * j + 1], njz = nrm[3 * j + 2];
    float4 f;
    f.x = safe_norm3(px, py, pz);
    f.y = angle3(nix, niy, niz, px, py, pz);
    f.z = angle3(njx, njy, njz, px, py, pz);
    f.w = angle3(nix, niy, niz, njx, njy, njz);
    ((float4*)feats)[e] = f;
}

// ---------------------------------------------------------------- conv1
__global__ void conv1_kernel(const float* __restrict__ feats,
                             const _Float16* __restrict__ packedW,
                             const float* __restrict__ b1a,
                             const float* __restrict__ b1b,
                             float* __restrict__ x1) {
    __shared__ _Float16 hid[WAVES_PER_BLOCK][16][HID];
    const int lane = threadIdx.x & 31;
    const int wid  = threadIdx.x >> 5;
    const int eb   = (blockIdx.x * WAVES_PER_BLOCK + wid) * 16;   // EE % 16 == 0
    const int m    = lane & 15;
    const bool hiK = lane >= 16;
    const int edge = eb + m;
    const int col  = lane & 15;

    // A (16x32 f16): only K=0..3 nonzero -> only lanes<16, elems 0..3
    v16h a = zero_v16h();
    if (!hiK) {
        const float4 f = ((const float4*)feats)[edge];            // one b128
        a[0] = (_Float16)f.x; a[1] = (_Float16)f.y;
        a[2] = (_Float16)f.z; a[3] = (_Float16)f.w;
    }
    v8f d0 = wmma16(a, load_Bp(packedW, 0, lane), bias_c(b1a, col));
    v8f d1 = wmma16(a, load_Bp(packedW, 1, lane), bias_c(b1a, col + 16));

    const int rowoff = hiK ? 8 : 0;
#pragma unroll
    for (int r = 0; r < 8; ++r) {
        hid[wid][r + rowoff][col]      = (_Float16)fmaxf(d0[r], 0.0f);
        hid[wid][r + rowoff][col + 16] = (_Float16)fmaxf(d1[r], 0.0f);
    }
    __syncthreads();

    // A2: row m, K layout {0..7,16..23} / {8..15,24..31}
    v16h a2;
    const int kb = hiK ? 8 : 0;
#pragma unroll
    for (int t = 0; t < 8; ++t) {
        a2[t]     = hid[wid][m][kb + t];
        a2[t + 8] = hid[wid][m][16 + kb + t];
    }
    v8f o0 = wmma16(a2, load_Bp(packedW, 2, lane), bias_c(b1b, col));
    v8f o1 = wmma16(a2, load_Bp(packedW, 3, lane), bias_c(b1b, col + 16));

    // relu(segment_max) == int atomicMax of clamped (>=0) values on 0-init buf
#pragma unroll
    for (int r = 0; r < 8; ++r) {
        const int ed  = eb + r + rowoff;
        const int dst = ed / EPN;
        atomicMax((int*)&x1[dst * HID + col],      __float_as_int(fmaxf(o0[r], 0.0f)));
        atomicMax((int*)&x1[dst * HID + col + 16], __float_as_int(fmaxf(o1[r], 0.0f)));
    }
}

// ---------------------------------------------------------------- conv2
__global__ void conv2_kernel(const float* __restrict__ feats,
                             const int* __restrict__ nbr,
                             const float* __restrict__ x1,
                             const _Float16* __restrict__ packedW,
                             const float* __restrict__ b2a,
                             const float* __restrict__ b2b,
                             float* __restrict__ x2) {
    __shared__ _Float16 hid[WAVES_PER_BLOCK][16][HID];
    const int lane = threadIdx.x & 31;
    const int wid  = threadIdx.x >> 5;
    const int eb   = (blockIdx.x * WAVES_PER_BLOCK + wid) * 16;
    const int m    = lane & 15;
    const bool hiK = lane >= 16;
    const int edge = eb + m;
    const int i    = edge / EPN;
    const int k    = edge - i * EPN;
    const int src  = (k < KNN) ? nbr[i * KNN + k] : i;
    const int col  = lane & 15;
    const int kb   = hiK ? 8 : 0;

    // K-chunk 0: x1[src][0..31]; 8 consecutive floats per half -> 4x b128
    v16h a0;
    {
        const float4* xp = (const float4*)(x1 + (size_t)src * HID);
        const int q = kb >> 2;                         // 0 or 2
        const float4 u0 = xp[q],     u1 = xp[q + 1];
        const float4 u2 = xp[q + 4], u3 = xp[q + 5];
        a0[0]  = (_Float16)u0.x; a0[1]  = (_Float16)u0.y;
        a0[2]  = (_Float16)u0.z; a0[3]  = (_Float16)u0.w;
        a0[4]  = (_Float16)u1.x; a0[5]  = (_Float16)u1.y;
        a0[6]  = (_Float16)u1.z; a0[7]  = (_Float16)u1.w;
        a0[8]  = (_Float16)u2.x; a0[9]  = (_Float16)u2.y;
        a0[10] = (_Float16)u2.z; a0[11] = (_Float16)u2.w;
        a0[12] = (_Float16)u3.x; a0[13] = (_Float16)u3.y;
        a0[14] = (_Float16)u3.z; a0[15] = (_Float16)u3.w;
    }
    // K-chunk 1: inputs 32..35 = ppf feats, rest zero
    v16h a1 = zero_v16h();
    if (!hiK) {
        const float4 f = ((const float4*)feats)[edge];
        a1[0] = (_Float16)f.x; a1[1] = (_Float16)f.y;
        a1[2] = (_Float16)f.z; a1[3] = (_Float16)f.w;
    }
    v8f d0 = wmma16(a0, load_Bp(packedW, 4, lane), bias_c(b2a, col));
    d0     = wmma16(a1, load_Bp(packedW, 6, lane), d0);
    v8f d1 = wmma16(a0, load_Bp(packedW, 5, lane), bias_c(b2a, col + 16));
    d1     = wmma16(a1, load_Bp(packedW, 7, lane), d1);

    const int rowoff = hiK ? 8 : 0;
#pragma unroll
    for (int r = 0; r < 8; ++r) {
        hid[wid][r + rowoff][col]      = (_Float16)fmaxf(d0[r], 0.0f);
        hid[wid][r + rowoff][col + 16] = (_Float16)fmaxf(d1[r], 0.0f);
    }
    __syncthreads();

    v16h a2;
#pragma unroll
    for (int t = 0; t < 8; ++t) {
        a2[t]     = hid[wid][m][kb + t];
        a2[t + 8] = hid[wid][m][16 + kb + t];
    }
    v8f o0 = wmma16(a2, load_Bp(packedW, 8, lane), bias_c(b2b, col));
    v8f o1 = wmma16(a2, load_Bp(packedW, 9, lane), bias_c(b2b, col + 16));

#pragma unroll
    for (int r = 0; r < 8; ++r) {
        const int ed  = eb + r + rowoff;
        const int dst = ed / EPN;
        atomicMax((int*)&x2[dst * HID + col],      __float_as_int(fmaxf(o0[r], 0.0f)));
        atomicMax((int*)&x2[dst * HID + col + 16], __float_as_int(fmaxf(o1[r], 0.0f)));
    }
}

// ---------------------------------------------------------------- pool + fc
__global__ void pool_fc_kernel(const float* __restrict__ x2,
                               const float* __restrict__ wc,
                               const float* __restrict__ bc,
                               float* __restrict__ out) {
    __shared__ float red[256];
    __shared__ float gmax[HID];
    const int g = blockIdx.x;
    const int t = threadIdx.x;
    const int c = t & 31;
    const int part = t >> 5;                           // 0..7
    float mx = 0.0f;                                   // values are >= 0 (post-relu)
    for (int n = part; n < NN; n += 8)
        mx = fmaxf(mx, x2[((g << 12) + n) * HID + c]);
    red[t] = mx;
    __syncthreads();
    if (t < HID) {
        float v = red[t];
#pragma unroll
        for (int p = 1; p < 8; ++p) v = fmaxf(v, red[t + (p << 5)]);
        gmax[t] = v;
    }
    __syncthreads();
    if (t < NCLS) {
        float acc = bc[t];
#pragma unroll
        for (int cc = 0; cc < HID; ++cc) acc += gmax[cc] * wc[cc * NCLS + t];
        out[g * NCLS + t] = acc;
    }
}

// ---------------------------------------------------------------- launch
extern "C" void kernel_launch(void* const* d_in, const int* in_sizes, int n_in,
                              void* d_out, int out_size, void* d_ws, size_t ws_size,
                              hipStream_t stream) {
    const float* pos = (const float*)d_in[0];
    const float* nrm = (const float*)d_in[1];
    // d_in[2] = batch (unused: graphs are uniform, size N each)
    const float* w1a = (const float*)d_in[3];
    const float* b1a = (const float*)d_in[4];
    const float* w1b = (const float*)d_in[5];
    const float* b1b = (const float*)d_in[6];
    const float* w2a = (const float*)d_in[7];
    const float* b2a = (const float*)d_in[8];
    const float* w2b = (const float*)d_in[9];
    const float* b2b = (const float*)d_in[10];
    const float* wc  = (const float*)d_in[11];
    const float* bc  = (const float*)d_in[12];
    float* out = (float*)d_out;

    char* ws = (char*)d_ws;
    const size_t nbrBytes  = (size_t)MM * KNN * sizeof(int);        // 2 MB
    const size_t featBytes = (size_t)EE * 4 * sizeof(float);        // 8.9 MB
    const size_t xBytes    = (size_t)MM * HID * sizeof(float);      // 4 MB
    int*      nbr   = (int*)ws;
    float*    feats = (float*)(ws + nbrBytes);
    float*    x1    = (float*)(ws + nbrBytes + featBytes);
    float*    x2    = (float*)(ws + nbrBytes + featBytes + xBytes);
    _Float16* pw    = (_Float16*)(ws + nbrBytes + featBytes + 2 * xBytes);

    hipMemsetAsync(x1, 0, xBytes, stream);
    hipMemsetAsync(x2, 0, xBytes, stream);

    pack_weights_kernel<<<1, NTILES * 32, 0, stream>>>(w1a, w1b, w2a, w2b, pw);
    knn_kernel<<<MM / 256, 256, 0, stream>>>(pos, nbr);
    ppf_kernel<<<EE / 256, 256, 0, stream>>>(pos, nrm, nbr, feats);
    const int convBlocks = EE / 16 / WAVES_PER_BLOCK;               // 4352, exact
    conv1_kernel<<<convBlocks, 256, 0, stream>>>(feats, pw, b1a, b1b, x1);
    conv2_kernel<<<convBlocks, 256, 0, stream>>>(feats, nbr, x1, pw, b2a, b2b, x2);
    pool_fc_kernel<<<BGR, 256, 0, stream>>>(x2, wc, bc, out);
}
